// CR8_reg_cond_mul_2_13975823582039
// MI455X (gfx1250) — matrix operand order repaired and checked by reading, not compiled, not run on hardware
//
#include <hip/hip_runtime.h>

typedef __attribute__((ext_vector_type(16))) __bf16 v16bf;
typedef __attribute__((ext_vector_type(8)))  float  v8f;

#define PITCH 144          // bf16 elements per LDS row (288B: 16B-aligned, bank-rotating)
#define SLOPE 0.01f

union Frag16 { uint4 q[2]; v16bf v; };
union Pack8  { uint4 q;    __bf16 h[8]; };

__device__ __forceinline__ float lrelu_f(float x) { return x >= 0.f ? x : SLOPE * x; }

// Y[128 out][128 pos] = W(128x128 f32, row-major OxC) * X (LDS [pos][ch] bf16).
// Epilogue: y = acc * (HAS_SCALE ? scale[ch] : 1) + off[ch]; optional leaky relu.
// Store bf16 into Yl [pos][ch]. Flags are compile-time so no branches survive.
template<bool HAS_SCALE, bool RELU>
__device__ __forceinline__ void gemm128(
    const float* __restrict__ Wg,
    const __bf16* __restrict__ Xl,
    __bf16* __restrict__ Yl,
    const float* __restrict__ scale,   // used iff HAS_SCALE
    const float* __restrict__ off,     // always valid (LDS or global)
    int wave, int lane)
{
  const int m  = lane & 15;   // A row within tile / B column (pos) within tile
  const int kh = lane >> 4;   // half-of-K selector per ISA layouts

  // A fragments: out channels [wave*16 .. wave*16+15], K = 32k..32k+31.
  // 16-bit A 16x32 (contiguous K, matching B): lane<16 -> elems 0..15 = K 0..15;
  //                     lane>=16 -> K 16..31 (aligned 16B runs of f32x4).
  v16bf A[4];
  const int row = wave * 16 + m;
  #pragma unroll
  for (int k = 0; k < 4; ++k) {
    const float* wp = Wg + row * 128 + 32 * k + kh * 16;
    const float4 fa = ((const float4*)wp)[0];
    const float4 fb = ((const float4*)wp)[1];
    const float4 fc = ((const float4*)(wp + 8))[0];
    const float4 fd = ((const float4*)(wp + 8))[1];
    Frag16 u;
    u.v[0]  = (__bf16)fa.x; u.v[1]  = (__bf16)fa.y; u.v[2]  = (__bf16)fa.z; u.v[3]  = (__bf16)fa.w;
    u.v[4]  = (__bf16)fb.x; u.v[5]  = (__bf16)fb.y; u.v[6]  = (__bf16)fb.z; u.v[7]  = (__bf16)fb.w;
    u.v[8]  = (__bf16)fc.x; u.v[9]  = (__bf16)fc.y; u.v[10] = (__bf16)fc.z; u.v[11] = (__bf16)fc.w;
    u.v[12] = (__bf16)fd.x; u.v[13] = (__bf16)fd.y; u.v[14] = (__bf16)fd.z; u.v[15] = (__bf16)fd.w;
    A[k] = u.v;
  }

  // Per-lane epilogue constants: D-layout gives this lane channels chbase..chbase+7
  // for every N-tile, so load scale/off once (contiguous 32B runs -> vector loads).
  const int chbase = wave * 16 + 8 * kh;
  float s8[8], o8[8];
  #pragma unroll
  for (int r = 0; r < 8; ++r) {
    s8[r] = HAS_SCALE ? scale[chbase + r] : 1.0f;
    o8[r] = off[chbase + r];
  }

  #pragma unroll
  for (int nt = 0; nt < 8; ++nt) {
    v8f acc = {};
    const int pos = nt * 16 + m;
    #pragma unroll
    for (int k = 0; k < 4; ++k) {
      // 16-bit B 32x16 (contiguous K, matching A): lanes 0-15 hold K=0..15 (col=lane), lanes 16-31 K=16..31.
      Frag16 bu;
      const uint4* q = (const uint4*)(Xl + pos * PITCH + 32 * k + kh * 16);
      bu.q[0] = q[0]; bu.q[1] = q[1];
      acc = __builtin_amdgcn_wmma_f32_16x16x32_bf16(
          false, A[k], false, bu.v, (short)0, acc, false, false);
    }
    // D layout: lane holds N = lane&15 (pos), M = r + 8*(lane>>4).
    Pack8 st;
    #pragma unroll
    for (int r = 0; r < 8; ++r) {
      float y = acc[r] * s8[r] + o8[r];
      if (RELU) y = lrelu_f(y);
      st.h[r] = (__bf16)y;
    }
    *(uint4*)(Yl + pos * PITCH + chbase) = st.q;
  }
}

__global__ __launch_bounds__(256)
void fused_seg_reg_kernel(
    const float* __restrict__ x_in,
    const float* __restrict__ cl1_w, const float* __restrict__ cl1_b,
    const float* __restrict__ bn1_g, const float* __restrict__ bn1_be,
    const float* __restrict__ bn1_m, const float* __restrict__ bn1_v,
    const float* __restrict__ cl2_w, const float* __restrict__ cl2_b,
    const float* __restrict__ cl3_w, const float* __restrict__ cl3_b,
    const float* __restrict__ reg1_w, const float* __restrict__ reg1_b,
    const float* __restrict__ bnr_g, const float* __restrict__ bnr_be,
    const float* __restrict__ bnr_m, const float* __restrict__ bnr_v,
    const float* __restrict__ cm2_w, const float* __restrict__ cm2_b,
    const float* __restrict__ cm3_w, const float* __restrict__ cm3_b,
    float* __restrict__ out)
{
  __shared__ __attribute__((aligned(16))) __bf16 xb[128 * PITCH]; // input tile, later logits
  __shared__ __attribute__((aligned(16))) __bf16 t1[128 * PITCH]; // cl1 out, later xr
  __shared__ __attribute__((aligned(16))) __bf16 t2[128 * PITCH]; // cl2 out
  __shared__ float s1[128], o1[128], sR[128], oR[128];
  __shared__ int   indl[128];
  __shared__ float partl[256];

  const int t    = threadIdx.x;
  const int lane = t & 31;
  const int wave = t >> 5;
  const int b    = blockIdx.x >> 6;          // 64 tiles of 128 positions per batch
  const int w0   = (blockIdx.x & 63) * 128;
  const size_t pbase = (size_t)b * 8192 + w0;

  // ---- Stage 0: load x_in tile -> bf16 LDS [pos][ch]; fold BN params ----
  {
    const int pos = t & 127, half = t >> 7;
    const float* xi = x_in + ((size_t)b * 128 + half * 64) * 8192 + w0 + pos;
    #pragma unroll 4
    for (int c = 0; c < 64; ++c)
      xb[pos * PITCH + half * 64 + c] = (__bf16)xi[(size_t)c * 8192];
    if (half == 0) {
      const float sc = bn1_g[pos] * rsqrtf(bn1_v[pos] + 1e-5f);
      s1[pos] = sc;
      o1[pos] = (cl1_b[pos] - bn1_m[pos]) * sc + bn1_be[pos];
    } else {
      const float sc = bnr_g[pos] * rsqrtf(bnr_v[pos] + 1e-5f);
      sR[pos] = sc;
      oR[pos] = (reg1_b[pos] - bnr_m[pos]) * sc + bnr_be[pos];
    }
  }
  __syncthreads();

  // ---- WMMA GEMM chain (each: 8 waves x 32 v_wmma_f32_16x16x32_bf16) ----
  gemm128<true,  true >(cl1_w,  xb, t1, s1,      o1,    wave, lane);  // x1
  __syncthreads();
  gemm128<false, true >(cl2_w,  t1, t2, nullptr, cl2_b, wave, lane);  // x2
  __syncthreads();
  gemm128<true,  true >(reg1_w, xb, t1, sR,      oR,    wave, lane);  // xr (t1)
  __syncthreads();
  gemm128<false, false>(cl3_w,  t2, xb, nullptr, cl3_b, wave, lane);  // logits (xb)
  __syncthreads();

  // ---- mask (channel 128 of cl3) and argmax over logits ----
  if (t < 128) {
    const float* w3r = cl3_w + 128 * 128;   // last row of (129,128)
    float acc = 0.f;
    #pragma unroll 4
    for (int c = 0; c < 128; ++c) acc += (float)t2[t * PITCH + c] * w3r[c];
    acc += cl3_b[128];
    out[65536 + pbase + t] = lrelu_f(acc);
  } else {
    const int pos = t - 128;
    float best = -3.4e38f; int bi = 0;
    for (int c = 0; c < 128; ++c) {          // first-max-index semantics
      const float v = (float)xb[pos * PITCH + c];
      if (v > best) { best = v; bi = c; }
    }
    indl[pos] = bi;
  }
  __syncthreads();

  // ---- gathered per-position MLP: 128 -> 32 (lrelu) -> 1 ----
  {
    const int pos = t & 127, half = t >> 7;  // 2 threads/pos, 16 outputs each
    const int ind = indl[pos];
    const float4* w2 = (const float4*)(cm2_w + (size_t)ind * 4096 + half * 16);
    float acc[16];
    #pragma unroll
    for (int o = 0; o < 16; ++o) acc[o] = 0.f;
    for (int c = 0; c < 128; ++c) {
      const float xv = (float)t1[pos * PITCH + c];
      const float4* wr = w2 + c * 8;         // 32 floats/row = 8 float4
      #pragma unroll
      for (int q4 = 0; q4 < 4; ++q4) {
        const float4 wv = wr[q4];
        acc[q4 * 4 + 0] += xv * wv.x;
        acc[q4 * 4 + 1] += xv * wv.y;
        acc[q4 * 4 + 2] += xv * wv.z;
        acc[q4 * 4 + 3] += xv * wv.w;
      }
    }
    const float* b2 = cm2_b + ind * 32 + half * 16;
    const float* w3 = cm3_w + ind * 32 + half * 16;
    float part = 0.f;
    #pragma unroll
    for (int o = 0; o < 16; ++o) {
      const float h = lrelu_f(acc[o] + b2[o]);
      part += h * w3[o];
    }
    partl[half * 128 + pos] = part;
  }
  __syncthreads();

  if (t < 128) {
    const int ind = indl[t];
    const float reg = partl[t] + partl[128 + t] + cm3_b[ind];
    out[pbase + t] = ((float)ind + reg) * (1.0f / 128.0f);
  }
}

extern "C" void kernel_launch(void* const* d_in, const int* in_sizes, int n_in,
                              void* d_out, int out_size, void* d_ws, size_t ws_size,
                              hipStream_t stream) {
  (void)in_sizes; (void)n_in; (void)d_ws; (void)ws_size; (void)out_size;
  const float* x_in  = (const float*)d_in[0];
  const float* cl1_w = (const float*)d_in[1];
  const float* cl1_b = (const float*)d_in[2];
  const float* bn1_g = (const float*)d_in[3];
  const float* bn1_be= (const float*)d_in[4];
  const float* bn1_m = (const float*)d_in[5];
  const float* bn1_v = (const float*)d_in[6];
  const float* cl2_w = (const float*)d_in[7];
  const float* cl2_b = (const float*)d_in[8];
  const float* cl3_w = (const float*)d_in[9];
  const float* cl3_b = (const float*)d_in[10];
  const float* reg1_w= (const float*)d_in[11];
  const float* reg1_b= (const float*)d_in[12];
  const float* bnr_g = (const float*)d_in[13];
  const float* bnr_be= (const float*)d_in[14];
  const float* bnr_m = (const float*)d_in[15];
  const float* bnr_v = (const float*)d_in[16];
  const float* cm2_w = (const float*)d_in[17];
  const float* cm2_b = (const float*)d_in[18];
  const float* cm3_w = (const float*)d_in[19];
  const float* cm3_b = (const float*)d_in[20];
  float* out = (float*)d_out;

  dim3 grid(512), block(256);   // 512 tiles of 128 positions; 8 wave32 per block
  hipLaunchKernelGGL(fused_seg_reg_kernel, grid, block, 0, stream,
                     x_in, cl1_w, cl1_b, bn1_g, bn1_be, bn1_m, bn1_v,
                     cl2_w, cl2_b, cl3_w, cl3_b, reg1_w, reg1_b,
                     bnr_g, bnr_be, bnr_m, bnr_v, cm2_w, cm2_b, cm3_w, cm3_b,
                     out);
}
